// WaveNetBackbone_73675868996098
// MI455X (gfx1250) — matrix operand order repaired and checked by reading, not compiled, not run on hardware
//
#include <hip/hip_runtime.h>

typedef __bf16 bf16_t;
typedef __attribute__((ext_vector_type(16))) __bf16 v16bf;
typedef __attribute__((ext_vector_type(8)))  float  v8f;

#define CC 128          // channels
#define TT 64           // time tile per workgroup
#define WSTRIDE 136     // padded LDS row stride (bf16 elems): 272B = 68 banks -> conflict-free
#define ASTRIDE 136

// ---------------- fragment loaders (layouts per CDNA5 ISA 7.12.2) ----------------

// A-matrix 16x32 bf16: lane<16 holds row M=lane, K={kk*32+0..7, kk*32+16..23};
// lane>=16 holds row M=lane-16, K={kk*32+8..15, kk*32+24..31}.
__device__ inline v16bf load_frag_a(const bf16_t* w, int co_base, int kk, int lane) {
    const int m  = lane & 15;
    const int kh = lane >> 4;
    const bf16_t* p = w + (co_base + m) * WSTRIDE + kk * 32 + kh * 8;
    union { v16bf v; uint4 u[2]; } x;
    x.u[0] = *(const uint4*)p;         // first 8 K
    x.u[1] = *(const uint4*)(p + 16);  // second 8 K (offset +16 in K)
    return x.v;
}

// B-matrix 32x16 bf16: lane<16 = column N=lane, K={kk*32+0..7, kk*32+16..23};
// lane>=16 = column N=lane-16, K={kk*32+8..15, kk*32+24..31}.
__device__ inline v16bf load_frag_b(const bf16_t* act, int t_base, int kk, int lane) {
    const int n  = lane & 15;
    const int kh = lane >> 4;
    const bf16_t* p = act + (t_base + n) * ASTRIDE + kk * 32 + kh * 8;
    union { v16bf v; uint4 u[2]; } x;
    x.u[0] = *(const uint4*)p;
    x.u[1] = *(const uint4*)(p + 16);
    return x.v;
}

__device__ inline v8f wmma_bf16(v16bf a, v16bf b, v8f c) {
    return __builtin_amdgcn_wmma_f32_16x16x32_bf16(false, a, false, b, (short)0, c, false, false);
}

// pack 8 f32 accumulator elements (contiguous co) -> 8 bf16, one 16B store
__device__ inline void store_bf8(bf16_t* p, v8f v) {
    union { bf16_t h[8]; uint4 u; } x;
#pragma unroll
    for (int i = 0; i < 8; ++i) x.h[i] = (bf16_t)v[i];
    *(uint4*)p = x.u;
}

// copy 128x128 bf16 weight matrix global->LDS with row padding
__device__ inline void load_w_lds(bf16_t* dst, const bf16_t* src, int tid) {
    const int row = tid >> 1, half = tid & 1;
    const uint4* s = (const uint4*)(src + row * CC + half * 64);
    uint4*       d = (uint4*)(dst + row * WSTRIDE + half * 64);
#pragma unroll
    for (int i = 0; i < 8; ++i) d[i] = s[i];
}

__device__ inline void gemm_tile(const bf16_t* w, const bf16_t* act, int co_base, int lane, v8f acc[4]) {
#pragma unroll
    for (int kk = 0; kk < 4; ++kk) {
        v16bf a = load_frag_a(w, co_base, kk, lane);
#pragma unroll
        for (int ct = 0; ct < 4; ++ct) {
            v16bf b = load_frag_b(act, ct * 16, kk, lane);
            acc[ct] = wmma_bf16(a, b, acc[ct]);
        }
    }
}

// fused dual GEMM sharing B fragments (tanh & sigmoid branches)
__device__ inline void gemm_tile_dual(const bf16_t* wA, const bf16_t* wB, const bf16_t* act,
                                      int co_base, int lane, v8f aU[4], v8f aV[4]) {
#pragma unroll
    for (int kk = 0; kk < 4; ++kk) {
        v16bf at = load_frag_a(wA, co_base, kk, lane);
        v16bf as = load_frag_a(wB, co_base, kk, lane);
#pragma unroll
        for (int ct = 0; ct < 4; ++ct) {
            v16bf b = load_frag_b(act, ct * 16, kk, lane);
            aU[ct] = wmma_bf16(at, b, aU[ct]);
            aV[ct] = wmma_bf16(as, b, aV[ct]);
        }
    }
}

// ---------------- prep kernels ----------------

__global__ void prep_weights(const float* __restrict__ Wdil, const float* __restrict__ Wt,
                             const float* __restrict__ Ws,   const float* __restrict__ Wk,
                             bf16_t* wd0, bf16_t* wd1, bf16_t* wtb, bf16_t* wsb, bf16_t* wkb, int n) {
    int i = blockIdx.x * blockDim.x + threadIdx.x;
    if (i < n) {
        wd0[i] = (bf16_t)Wdil[2 * i];      // tap at t-dil
        wd1[i] = (bf16_t)Wdil[2 * i + 1];  // tap at t
        wtb[i] = (bf16_t)Wt[i];
        wsb[i] = (bf16_t)Ws[i];
        wkb[i] = (bf16_t)Wk[i];
    }
}

// h[b][t][c] = W_in[c] * x[b][t] + b_in[c]   (CIN = 1)
__global__ void input_proj(const float* __restrict__ x, const float* __restrict__ Win,
                           const float* __restrict__ bin, bf16_t* __restrict__ h, long n) {
    long idx = (long)blockIdx.x * blockDim.x + threadIdx.x;
    if (idx < n) {
        int  c  = (int)(idx & (CC - 1));
        long bt = idx >> 7;
        h[idx] = (bf16_t)(Win[c] * x[bt] + bin[c]);
    }
}

// skip [B][T][C] f32 -> out [B][C][T] f32
__global__ void transpose_out(const float* __restrict__ s, float* __restrict__ out, int T) {
    __shared__ float tile[32][33];
    const int b = blockIdx.z, t0 = blockIdx.x * 32, c0 = blockIdx.y * 32;
    const int tx = threadIdx.x, ty = threadIdx.y;
#pragma unroll
    for (int j = 0; j < 4; ++j) {
        int r = ty + j * 8;  // t-local
        tile[r][tx] = s[((size_t)b * T + t0 + r) * CC + c0 + tx];
    }
    __syncthreads();
#pragma unroll
    for (int j = 0; j < 4; ++j) {
        int r = ty + j * 8;  // c-local
        out[((size_t)b * CC + c0 + r) * T + t0 + tx] = tile[tx][r];
    }
}

// ---------------- fused WaveNet layer ----------------

__global__ __launch_bounds__(256) void wavenet_layer(
    const bf16_t* __restrict__ h_in, bf16_t* __restrict__ h_out, float* __restrict__ skip,
    const bf16_t* __restrict__ Wd0, const bf16_t* __restrict__ Wd1,
    const bf16_t* __restrict__ Wt,  const bf16_t* __restrict__ Ws, const bf16_t* __restrict__ Wk,
    const float* __restrict__ bd, const float* __restrict__ bt,
    const float* __restrict__ bs, const float* __restrict__ bk,
    int dil, int first, int T)
{
    __shared__ __align__(16) bf16_t lds_w[2 * CC * WSTRIDE];   // 68 KB, phase-reused
    __shared__ __align__(16) bf16_t lds_a[2 * TT * ASTRIDE];   // 34 KB, phase-reused
    bf16_t* w0 = lds_w;
    bf16_t* w1 = lds_w + CC * WSTRIDE;
    bf16_t* a0 = lds_a;                 // h[t-dil]  -> later xd (bf16)
    bf16_t* a1 = lds_a + TT * ASTRIDE;  // h[t]      -> later xh (bf16)

    const int tid  = threadIdx.x;
    const int lane = tid & 31;
    const int wave = tid >> 5;
    const int b    = blockIdx.y;
    const int t0   = blockIdx.x * TT;
    const int co_base = wave * 16;
    const int kh = lane >> 4;
    const int nn = lane & 15;
    const int cbase = co_base + 8 * kh;   // 8 contiguous co per lane in C/D layout

    // ---- phase A: stage Wd0/Wd1 + activation tiles ----
    load_w_lds(w0, Wd0, tid);
    load_w_lds(w1, Wd1, tid);
    {
        const int row = tid >> 2, part = tid & 3;
        const int tg = t0 - dil + row;                     // causal pad with zeros
        uint4* d0 = (uint4*)(a0 + row * ASTRIDE) + part * 4;
        if (tg >= 0) {
            const uint4* s0 = (const uint4*)(h_in + ((size_t)b * T + tg) * CC) + part * 4;
#pragma unroll
            for (int i = 0; i < 4; ++i) d0[i] = s0[i];
        } else {
            uint4 z = {0u, 0u, 0u, 0u};
#pragma unroll
            for (int i = 0; i < 4; ++i) d0[i] = z;
        }
        const uint4* s1 = (const uint4*)(h_in + ((size_t)b * T + t0 + row) * CC) + part * 4;
        uint4* d1 = (uint4*)(a1 + row * ASTRIDE) + part * 4;
#pragma unroll
        for (int i = 0; i < 4; ++i) d1[i] = s1[i];
    }
    __syncthreads();

    // ---- GEMM1: xd = Wd0 @ h[t-dil] + Wd1 @ h[t] + bd ----
    const v8f vzero = {0.f, 0.f, 0.f, 0.f, 0.f, 0.f, 0.f, 0.f};
    v8f accXD[4] = {vzero, vzero, vzero, vzero};
    gemm_tile(w0, a0, co_base, lane, accXD);
    gemm_tile(w1, a1, co_base, lane, accXD);
#pragma unroll
    for (int ct = 0; ct < 4; ++ct)
#pragma unroll
        for (int i = 0; i < 8; ++i) accXD[ct][i] += bd[cbase + i];

    __syncthreads();  // all reads of w0/w1/a0/a1 done -> safe to overwrite

    // ---- stage xd (bf16) into a0; stage Wt/Ws over Wd ----
#pragma unroll
    for (int ct = 0; ct < 4; ++ct)
        store_bf8(a0 + (ct * 16 + nn) * ASTRIDE + cbase, accXD[ct]);
    load_w_lds(w0, Wt, tid);
    load_w_lds(w1, Ws, tid);
    __syncthreads();

    // ---- GEMM2/3 fused: U = Wt@xd, V = Ws@xd ----
    v8f accU[4] = {vzero, vzero, vzero, vzero};
    v8f accV[4] = {vzero, vzero, vzero, vzero};
    gemm_tile_dual(w0, w1, a0, co_base, lane, accU, accV);

    // ---- gate + residual:  xh = tanh(U+bt)*sigmoid(V+bs);  h_new = xh + xd ----
#pragma unroll
    for (int ct = 0; ct < 4; ++ct) {
#pragma unroll
        for (int i = 0; i < 8; ++i) {
            float u = accU[ct][i] + bt[cbase + i];
            float v = accV[ct][i] + bs[cbase + i];
            float f = tanhf(u);
            float g = 1.f / (1.f + expf(-v));
            float xh = f * g;
            accU[ct][i]  = xh;                     // keep xh
            accXD[ct][i] = xh + accXD[ct][i];      // h_new
        }
        // write residual h_new (bf16) to global [B][T][C]
        store_bf8(h_out + ((size_t)b * T + t0 + ct * 16 + nn) * CC + cbase, accXD[ct]);
    }

    __syncthreads();  // all reads of w0/w1 (Wt/Ws) done -> safe to overwrite

    // ---- stage xh (bf16) into a1; stage Wk over Wt ----
#pragma unroll
    for (int ct = 0; ct < 4; ++ct)
        store_bf8(a1 + (ct * 16 + nn) * ASTRIDE + cbase, accU[ct]);
    load_w_lds(w0, Wk, tid);
    __syncthreads();

    // ---- GEMM4: skip += Wk @ xh + bk ----
    v8f accS[4] = {vzero, vzero, vzero, vzero};
    gemm_tile(w0, a1, co_base, lane, accS);
#pragma unroll
    for (int ct = 0; ct < 4; ++ct) {
        float* sp = skip + ((size_t)b * T + t0 + ct * 16 + nn) * CC + cbase;
        if (first) {
#pragma unroll
            for (int i = 0; i < 8; ++i) sp[i] = accS[ct][i] + bk[cbase + i];
        } else {
#pragma unroll
            for (int i = 0; i < 8; ++i) sp[i] += accS[ct][i] + bk[cbase + i];
        }
    }
}

// ---------------- host launcher ----------------

extern "C" void kernel_launch(void* const* d_in, const int* in_sizes, int n_in,
                              void* d_out, int out_size, void* d_ws, size_t ws_size,
                              hipStream_t stream) {
    (void)in_sizes; (void)n_in; (void)out_size; (void)ws_size;
    const int Bn = 8, T = 16384, Lr = 20;

    const float* x      = (const float*)d_in[0];
    const float* W_in   = (const float*)d_in[1];
    const float* b_in   = (const float*)d_in[2];
    const float* W_dil  = (const float*)d_in[3];
    const float* b_dil  = (const float*)d_in[4];
    const float* W_tanh = (const float*)d_in[5];
    const float* b_tanh = (const float*)d_in[6];
    const float* W_sig  = (const float*)d_in[7];
    const float* b_sig  = (const float*)d_in[8];
    const float* W_skip = (const float*)d_in[9];
    const float* b_skip = (const float*)d_in[10];

    // workspace layout (total ~137.5 MB)
    char* ws = (char*)d_ws;
    const size_t actBytes = (size_t)Bn * T * CC * 2;   // 32 MB bf16
    bf16_t* hA   = (bf16_t*)ws;
    bf16_t* hB   = (bf16_t*)(ws + actBytes);
    float*  skip = (float*)(ws + 2 * actBytes);        // 64 MB f32 [B][T][C]
    bf16_t* wbf  = (bf16_t*)(ws + 2 * actBytes + (size_t)Bn * T * CC * 4);
    const size_t wsz = (size_t)Lr * CC * CC;           // elems per weight array
    bf16_t* wd0 = wbf;
    bf16_t* wd1 = wbf + wsz;
    bf16_t* wt  = wbf + 2 * wsz;
    bf16_t* wsg = wbf + 3 * wsz;
    bf16_t* wk  = wbf + 4 * wsz;

    const int nW = Lr * CC * CC;
    prep_weights<<<(nW + 255) / 256, 256, 0, stream>>>(W_dil, W_tanh, W_sig, W_skip,
                                                       wd0, wd1, wt, wsg, wk, nW);

    const long nH = (long)Bn * T * CC;
    input_proj<<<(unsigned)((nH + 255) / 256), 256, 0, stream>>>(x, W_in, b_in, hA, nH);

    for (int l = 0; l < Lr; ++l) {
        const int dil = 1 << (l % 10);
        bf16_t* hi = (l & 1) ? hB : hA;
        bf16_t* ho = (l & 1) ? hA : hB;
        const size_t wo = (size_t)l * CC * CC;
        wavenet_layer<<<dim3(T / TT, Bn), 256, 0, stream>>>(
            hi, ho, skip,
            wd0 + wo, wd1 + wo, wt + wo, wsg + wo, wk + wo,
            b_dil + l * CC, b_tanh + l * CC, b_sig + l * CC, b_skip + l * CC,
            dil, (l == 0) ? 1 : 0, T);
    }

    transpose_out<<<dim3(T / 32, CC / 32, Bn), dim3(32, 8), 0, stream>>>(skip, (float*)d_out, T);
}
